// Self_Attention_57990648430953
// MI455X (gfx1250) — compile-verified, matrix operand order checked by reading, not run on hardware
//
#include <hip/hip_runtime.h>

// ---------------------------------------------------------------------------
// Self-attention w/ cosine-similarity bias, B=8, S=D=E=1024, fp32 in/out.
// All GEMMs on v_wmma_f32_16x16x32_bf16; BK=64 stages, double-buffered LDS,
// one s_barrier per stage, register staging to hide global latency.
// ---------------------------------------------------------------------------

typedef __attribute__((ext_vector_type(16))) __bf16 v16bf;
typedef __attribute__((ext_vector_type(8)))  float  v8f;

#define SEQ   1024
#define DIM   1024
#define NB    8
#define LDK2  72          // LDS row stride in bf16 elems (144B: 16B-aligned, bank-skewed)
#define TILE  (128 * LDK2)

union FragU { v16bf v; uint4 q[2]; };
union U4U   { uint4 q; unsigned short u[8]; };

__device__ __forceinline__ unsigned short f32_to_bf16(float f) {
  unsigned int u = __float_as_uint(f);
  u += 0x7FFFu + ((u >> 16) & 1u);       // round-to-nearest-even
  return (unsigned short)(u >> 16);
}
__device__ __forceinline__ float bf16_to_f32(unsigned short h) {
  return __uint_as_float((unsigned int)h << 16);
}

// 16x32 bf16 fragment from LDS tile. Lane map per ISA 7.12.2: row = lane&15;
// elems 0..7 -> k = kt + h*8 + 0..7, elems 8..15 -> k = kt + 16 + h*8 + 0..7.
__device__ __forceinline__ v16bf load_frag(const unsigned short* lds, int rc, int kt, int h) {
  FragU f;
  const unsigned short* p = lds + rc * LDK2 + kt + h * 8;
  f.q[0] = *(const uint4*)(p);
  f.q[1] = *(const uint4*)(p + 16);
  return f.v;
}

// One BK=64 stage: wave (wm 0..1, wn 0..3) owns a 64x32 tile of the 128x128
// block tile -> 4(m) x 2(n) WMMA tiles x 2 k-steps = 16 v_wmma per stage.
__device__ __forceinline__ void wave_mma64(const unsigned short* lds_a,
                                           const unsigned short* lds_b,
                                           int lane, int wm, int wn,
                                           v8f (&acc)[4][2]) {
  const int r = lane & 15, h = lane >> 4;
#pragma unroll
  for (int kt = 0; kt < 64; kt += 32) {
    v16bf bfr[2];
#pragma unroll
    for (int ni = 0; ni < 2; ++ni)
      bfr[ni] = load_frag(lds_b, wn * 32 + ni * 16 + r, kt, h);
#pragma unroll
    for (int mi = 0; mi < 4; ++mi) {
      v16bf afr = load_frag(lds_a, wm * 64 + mi * 16 + r, kt, h);
#pragma unroll
      for (int ni = 0; ni < 2; ++ni)
        acc[mi][ni] = __builtin_amdgcn_wmma_f32_16x16x32_bf16(
            false, afr, false, bfr[ni], (short)0, acc[mi][ni], false, false);
    }
  }
}

__device__ __forceinline__ void zero_acc(v8f (&acc)[4][2]) {
  const v8f z = {0.f, 0.f, 0.f, 0.f, 0.f, 0.f, 0.f, 0.f};
#pragma unroll
  for (int mi = 0; mi < 4; ++mi)
#pragma unroll
    for (int ni = 0; ni < 2; ++ni) acc[mi][ni] = z;
}

// ---------------- staging helpers (global -> regs -> LDS) -------------------
// A tile (row-major f32 source, 128 x 64): 2048 float4, 8 per thread.
__device__ __forceinline__ void ldg_a_f32(const float* src, int m0, int k0, int tid,
                                          float4 (&ra)[8]) {
#pragma unroll
  for (int i = 0; i < 8; ++i) {
    int idx = tid + i * 256, row = idx >> 4, c4 = idx & 15;
    ra[i] = *(const float4*)(src + (size_t)(m0 + row) * DIM + k0 + c4 * 4);
  }
}
__device__ __forceinline__ void sts_a_f32(unsigned short* pa, int tid, const float4 (&ra)[8]) {
#pragma unroll
  for (int i = 0; i < 8; ++i) {
    int idx = tid + i * 256, row = idx >> 4, c4 = idx & 15;
    unsigned short* d = pa + row * LDK2 + c4 * 4;
    d[0] = f32_to_bf16(ra[i].x); d[1] = f32_to_bf16(ra[i].y);
    d[2] = f32_to_bf16(ra[i].z); d[3] = f32_to_bf16(ra[i].w);
  }
}
// B tile transpose (f32 source [k][n] -> LDS B^T[n][k], 64 x 128): 8/thread.
__device__ __forceinline__ void ldg_b_f32(const float* src, int n0, int k0, int tid,
                                          float4 (&rb)[8]) {
#pragma unroll
  for (int i = 0; i < 8; ++i) {
    int idx = tid + i * 256, kk = idx >> 5, n4 = idx & 31;
    rb[i] = *(const float4*)(src + (size_t)(k0 + kk) * DIM + n0 + n4 * 4);
  }
}
__device__ __forceinline__ void sts_b_f32(unsigned short* pb, int tid, const float4 (&rb)[8]) {
#pragma unroll
  for (int i = 0; i < 8; ++i) {
    int idx = tid + i * 256, kk = idx >> 5, n4 = idx & 31;
    pb[(n4 * 4 + 0) * LDK2 + kk] = f32_to_bf16(rb[i].x);
    pb[(n4 * 4 + 1) * LDK2 + kk] = f32_to_bf16(rb[i].y);
    pb[(n4 * 4 + 2) * LDK2 + kk] = f32_to_bf16(rb[i].z);
    pb[(n4 * 4 + 3) * LDK2 + kk] = f32_to_bf16(rb[i].w);
  }
}
// Row-major bf16 tile (128 rows x 64 k): 1024 uint4, 4 per thread.
__device__ __forceinline__ void ldg_bf16_rm(const unsigned short* src, size_t boff, int r0,
                                            int k0, int ldm, int tid, uint4 (&rq)[4]) {
#pragma unroll
  for (int i = 0; i < 4; ++i) {
    int idx = tid + i * 256, row = idx >> 3, c = idx & 7;
    rq[i] = *(const uint4*)(src + boff + (size_t)(r0 + row) * ldm + k0 + c * 8);
  }
}
__device__ __forceinline__ void sts_bf16_rm(unsigned short* p, int tid, const uint4 (&rq)[4]) {
#pragma unroll
  for (int i = 0; i < 4; ++i) {
    int idx = tid + i * 256, row = idx >> 3, c = idx & 7;
    *(uint4*)(p + row * LDK2 + c * 8) = rq[i];
  }
}
// Transposed bf16 tile: LDS[n][k] = src[b][k0+kk][n0+n]; 64 kk x 128 n.
__device__ __forceinline__ void ldg_bf16_tr(const unsigned short* src, size_t boff, int n0,
                                            int k0, int tid, uint4 (&rq)[4]) {
#pragma unroll
  for (int i = 0; i < 4; ++i) {
    int idx = tid + i * 256, kk = idx >> 4, n8 = idx & 15;
    rq[i] = *(const uint4*)(src + boff + (size_t)(k0 + kk) * DIM + n0 + n8 * 8);
  }
}
__device__ __forceinline__ void sts_bf16_tr(unsigned short* p, int tid, const uint4 (&rq)[4]) {
#pragma unroll
  for (int i = 0; i < 4; ++i) {
    int idx = tid + i * 256, kk = idx >> 4, n8 = idx & 15;
    U4U d; d.q = rq[i];
#pragma unroll
    for (int j = 0; j < 8; ++j) p[(n8 * 8 + j) * LDK2 + kk] = d.u[j];
  }
}

// ---------------------------------------------------------------------------
// Kernel 1: QKV projections.  C[w][m][n] = sum_k x[m][k] * kern[w][k][n]
// grid = (64 mtiles, 8 ntiles, 3 weights), block = 256 (8 wave32)
// ---------------------------------------------------------------------------
__global__ __launch_bounds__(256) void qkv_kernel(
    const float* __restrict__ x, const float* __restrict__ kern,
    unsigned short* __restrict__ wq, unsigned short* __restrict__ wk,
    unsigned short* __restrict__ wv) {
  __shared__ __align__(16) unsigned short sa[2][TILE];
  __shared__ __align__(16) unsigned short sb[2][TILE];
  const int tid = threadIdx.x, lane = tid & 31, wid = tid >> 5;
  const int wm = wid & 1, wn = wid >> 1;
  const int m0 = blockIdx.x * 128, n0 = blockIdx.y * 128, w = blockIdx.z;
  const float* Bw = kern + (size_t)w * DIM * DIM;
  unsigned short* out = (w == 0) ? wq : (w == 1) ? wk : wv;

  v8f acc[4][2];
  zero_acc(acc);

  float4 ra[8], rb[8];
  ldg_a_f32(x, m0, 0, tid, ra);
  ldg_b_f32(Bw, n0, 0, tid, rb);
  sts_a_f32(sa[0], tid, ra);
  sts_b_f32(sb[0], tid, rb);
  __syncthreads();

  int cur = 0;
  for (int k0 = 64; k0 < DIM; k0 += 64) {
    if (k0 + 64 < DIM) {   // warm L2 for the tile after next
      __builtin_prefetch(x + (size_t)(m0 + (tid & 127)) * DIM + k0 + 64, 0, 1);
      __builtin_prefetch(Bw + (size_t)(k0 + 64 + (tid & 63)) * DIM + n0, 0, 1);
    }
    ldg_a_f32(x, m0, k0, tid, ra);
    ldg_b_f32(Bw, n0, k0, tid, rb);
    wave_mma64(sa[cur], sb[cur], lane, wm, wn, acc);
    sts_a_f32(sa[cur ^ 1], tid, ra);
    sts_b_f32(sb[cur ^ 1], tid, rb);
    cur ^= 1;
    __syncthreads();
  }
  wave_mma64(sa[cur], sb[cur], lane, wm, wn, acc);

  const int r = lane & 15, h = lane >> 4;
#pragma unroll
  for (int mi = 0; mi < 4; ++mi)
#pragma unroll
    for (int ni = 0; ni < 2; ++ni)
#pragma unroll
      for (int v = 0; v < 8; ++v) {
        int row = m0 + wm * 64 + mi * 16 + v + h * 8;
        int col = n0 + wn * 32 + ni * 16 + r;
        out[(size_t)row * DIM + col] = f32_to_bf16(acc[mi][ni][v]);
      }
}

// ---------------------------------------------------------------------------
// Kernel 2: L2 normalize WQ/WK along sequence axis (axis=1).
// ---------------------------------------------------------------------------
__global__ __launch_bounds__(256) void norm_kernel(
    const unsigned short* __restrict__ wq, const unsigned short* __restrict__ wk,
    unsigned short* __restrict__ nq, unsigned short* __restrict__ nk) {
  const int e = blockIdx.x * 256 + threadIdx.x;
  const int b = blockIdx.y;
  const unsigned short* src = blockIdx.z ? wk : wq;
  unsigned short*       dst = blockIdx.z ? nk : nq;
  const size_t base = (size_t)b * SEQ * DIM + e;
  float ss = 0.f;
  for (int s = 0; s < SEQ; ++s) {
    float v = bf16_to_f32(src[base + (size_t)s * DIM]);
    ss += v * v;
  }
  const float inv = rsqrtf(fmaxf(ss, 1e-12f));
  for (int s = 0; s < SEQ; ++s) {
    float v = bf16_to_f32(src[base + (size_t)s * DIM]);
    dst[base + (size_t)s * DIM] = f32_to_bf16(v * inv);
  }
}

// ---------------------------------------------------------------------------
// Kernel 3: scores[b][i][j] = 0.5*( (WQ[b,i,:].WK[b,j,:])/32
//                                  + sum_s nQ[b,s,i]*nK[b,s,j] )
// grid = (8 itiles, 8 jtiles, B)
// ---------------------------------------------------------------------------
__global__ __launch_bounds__(256) void scores_kernel(
    const unsigned short* __restrict__ wq, const unsigned short* __restrict__ wk,
    const unsigned short* __restrict__ nq, const unsigned short* __restrict__ nk,
    float* __restrict__ scores) {
  __shared__ __align__(16) unsigned short sa[2][TILE];
  __shared__ __align__(16) unsigned short sb[2][TILE];
  const int tid = threadIdx.x, lane = tid & 31, wid = tid >> 5;
  const int wm = wid & 1, wn = wid >> 1;
  const int i0 = blockIdx.x * 128, j0 = blockIdx.y * 128;
  const size_t boff = (size_t)blockIdx.z * SEQ * DIM;

  v8f acc1[4][2], acc2[4][2];
  zero_acc(acc1);
  zero_acc(acc2);

  uint4 qa[4], qb[4];

  // ---- pass 1: QK (contract over e); A/B^T are straight bf16 rows
  ldg_bf16_rm(wq, boff, i0, 0, DIM, tid, qa);
  ldg_bf16_rm(wk, boff, j0, 0, DIM, tid, qb);
  sts_bf16_rm(sa[0], tid, qa);
  sts_bf16_rm(sb[0], tid, qb);
  __syncthreads();
  int cur = 0;
  for (int e0 = 64; e0 < DIM; e0 += 64) {
    ldg_bf16_rm(wq, boff, i0, e0, DIM, tid, qa);
    ldg_bf16_rm(wk, boff, j0, e0, DIM, tid, qb);
    wave_mma64(sa[cur], sb[cur], lane, wm, wn, acc1);
    sts_bf16_rm(sa[cur ^ 1], tid, qa);
    sts_bf16_rm(sb[cur ^ 1], tid, qb);
    cur ^= 1;
    __syncthreads();
  }
  wave_mma64(sa[cur], sb[cur], lane, wm, wn, acc1);
  __syncthreads();

  // ---- pass 2: cosine (contract over s); transpose on LDS store
  ldg_bf16_tr(nq, boff, i0, 0, tid, qa);
  ldg_bf16_tr(nk, boff, j0, 0, tid, qb);
  sts_bf16_tr(sa[0], tid, qa);
  sts_bf16_tr(sb[0], tid, qb);
  __syncthreads();
  cur = 0;
  for (int s0 = 64; s0 < SEQ; s0 += 64) {
    ldg_bf16_tr(nq, boff, i0, s0, tid, qa);
    ldg_bf16_tr(nk, boff, j0, s0, tid, qb);
    wave_mma64(sa[cur], sb[cur], lane, wm, wn, acc2);
    sts_bf16_tr(sa[cur ^ 1], tid, qa);
    sts_bf16_tr(sb[cur ^ 1], tid, qb);
    cur ^= 1;
    __syncthreads();
  }
  wave_mma64(sa[cur], sb[cur], lane, wm, wn, acc2);

  const int r = lane & 15, h = lane >> 4;
  const float inv_sqrt_d = 0.03125f;  // 1/sqrt(1024)
#pragma unroll
  for (int mi = 0; mi < 4; ++mi)
#pragma unroll
    for (int ni = 0; ni < 2; ++ni)
#pragma unroll
      for (int v = 0; v < 8; ++v) {
        int row = i0 + wm * 64 + mi * 16 + v + h * 8;
        int col = j0 + wn * 32 + ni * 16 + r;
        scores[boff + (size_t)row * SEQ + col] =
            0.5f * (acc1[mi][ni][v] * inv_sqrt_d + acc2[mi][ni][v]);
      }
}

// ---------------------------------------------------------------------------
// Kernel 4: row softmax (axis=-1), f32 in -> bf16 probs out.
// ---------------------------------------------------------------------------
__global__ __launch_bounds__(256) void softmax_kernel(
    const float* __restrict__ scores, unsigned short* __restrict__ probs) {
  __shared__ float smax[8], ssum[8];
  const int tid = threadIdx.x, lane = tid & 31, wid = tid >> 5;
  const size_t base = (size_t)blockIdx.x * SEQ;
  float v[4];
  float m = -3.402823466e+38f;
#pragma unroll
  for (int i = 0; i < 4; ++i) {
    v[i] = scores[base + tid + i * 256];
    m = fmaxf(m, v[i]);
  }
#pragma unroll
  for (int off = 16; off > 0; off >>= 1) m = fmaxf(m, __shfl_xor(m, off, 32));
  if (lane == 0) smax[wid] = m;
  __syncthreads();
  if (tid == 0) {
    float t = smax[0];
#pragma unroll
    for (int i = 1; i < 8; ++i) t = fmaxf(t, smax[i]);
    smax[0] = t;
  }
  __syncthreads();
  m = smax[0];
  float s = 0.f;
#pragma unroll
  for (int i = 0; i < 4; ++i) {
    v[i] = __expf(v[i] - m);
    s += v[i];
  }
#pragma unroll
  for (int off = 16; off > 0; off >>= 1) s += __shfl_xor(s, off, 32);
  if (lane == 0) ssum[wid] = s;
  __syncthreads();
  if (tid == 0) {
    float t = 0.f;
#pragma unroll
    for (int i = 0; i < 8; ++i) t += ssum[i];
    ssum[0] = t;
  }
  __syncthreads();
  const float invs = 1.f / ssum[0];
#pragma unroll
  for (int i = 0; i < 4; ++i)
    probs[base + tid + i * 256] = f32_to_bf16(v[i] * invs);
}

// ---------------------------------------------------------------------------
// Kernel 5: out[b][i][e] = sum_t P[b][i][t] * WV[b][t][e], f32 output.
// grid = (8 itiles, 8 etiles, B)
// ---------------------------------------------------------------------------
__global__ __launch_bounds__(256) void out_kernel(
    const unsigned short* __restrict__ probs, const unsigned short* __restrict__ wv,
    float* __restrict__ out) {
  __shared__ __align__(16) unsigned short sa[2][TILE];
  __shared__ __align__(16) unsigned short sb[2][TILE];
  const int tid = threadIdx.x, lane = tid & 31, wid = tid >> 5;
  const int wm = wid & 1, wn = wid >> 1;
  const int i0 = blockIdx.x * 128, e0 = blockIdx.y * 128;
  const size_t boff = (size_t)blockIdx.z * SEQ * DIM;

  v8f acc[4][2];
  zero_acc(acc);

  uint4 qa[4], qb[4];
  ldg_bf16_rm(probs, boff, i0, 0, SEQ, tid, qa);
  ldg_bf16_tr(wv, boff, e0, 0, tid, qb);
  sts_bf16_rm(sa[0], tid, qa);
  sts_bf16_tr(sb[0], tid, qb);
  __syncthreads();

  int cur = 0;
  for (int t0 = 64; t0 < SEQ; t0 += 64) {
    ldg_bf16_rm(probs, boff, i0, t0, SEQ, tid, qa);
    ldg_bf16_tr(wv, boff, e0, t0, tid, qb);
    wave_mma64(sa[cur], sb[cur], lane, wm, wn, acc);
    sts_bf16_rm(sa[cur ^ 1], tid, qa);
    sts_bf16_tr(sb[cur ^ 1], tid, qb);
    cur ^= 1;
    __syncthreads();
  }
  wave_mma64(sa[cur], sb[cur], lane, wm, wn, acc);

  const int r = lane & 15, h = lane >> 4;
#pragma unroll
  for (int mi = 0; mi < 4; ++mi)
#pragma unroll
    for (int ni = 0; ni < 2; ++ni)
#pragma unroll
      for (int v = 0; v < 8; ++v) {
        int row = i0 + wm * 64 + mi * 16 + v + h * 8;
        int col = e0 + wn * 32 + ni * 16 + r;
        out[boff + (size_t)row * DIM + col] = acc[mi][ni][v];
      }
}

// ---------------------------------------------------------------------------
extern "C" void kernel_launch(void* const* d_in, const int* in_sizes, int n_in,
                              void* d_out, int out_size, void* d_ws, size_t ws_size,
                              hipStream_t stream) {
  const float* x    = (const float*)d_in[0];   // [8,1024,1024] f32
  const float* kern = (const float*)d_in[1];   // [3,1024,1024] f32
  float* out = (float*)d_out;                  // [8,1024,1024] f32

  const size_t NE = (size_t)NB * SEQ * DIM;    // 8388608 elems per tensor
  unsigned short* wq = (unsigned short*)d_ws;  // bf16, 16 MB each
  unsigned short* wk = wq + NE;
  unsigned short* wv = wk + NE;
  unsigned short* nq = wv + NE;
  unsigned short* nk = nq + NE;
  float* scores      = (float*)(nk + NE);      // 32 MB
  unsigned short* pr = (unsigned short*)(scores + NE);  // 16 MB

  qkv_kernel    <<<dim3(64, 8, 3), 256, 0, stream>>>(x, kern, wq, wk, wv);
  norm_kernel   <<<dim3(4, 8, 2),  256, 0, stream>>>(wq, wk, nq, nk);
  scores_kernel <<<dim3(8, 8, 8),  256, 0, stream>>>(wq, wk, nq, nk, scores);
  softmax_kernel<<<dim3(8192),     256, 0, stream>>>(scores, pr);
  out_kernel    <<<dim3(8, 8, 8),  256, 0, stream>>>(pr, wv, out);
}
